// BaiChuanAttention_5076651344290
// MI455X (gfx1250) — matrix operand order repaired and checked by reading, not compiled, not run on hardware
//
#include <hip/hip_runtime.h>
#include <hip/hip_bf16.h>

#define Bsz 2
#define Ssz 2048
#define Hsz 4096
#define NHsz 32
#define Dsz 128

typedef unsigned short ushort_t;
typedef __attribute__((ext_vector_type(16))) __bf16        bf16x16;
typedef __attribute__((ext_vector_type(8)))  float         f32x8;
typedef __attribute__((ext_vector_type(4)))  unsigned int  u32x4;

union Frag16 {
    bf16x16  v;
    ushort_t u[16];
    u32x4    q[2];
};

__device__ __forceinline__ ushort_t f2bf(float f) {
    unsigned u = __builtin_bit_cast(unsigned, f);
    u += 0x7FFFu + ((u >> 16) & 1u);     // round-to-nearest-even
    return (ushort_t)(u >> 16);
}
__device__ __forceinline__ float bf2f(ushort_t u) {
    return __builtin_bit_cast(float, (unsigned)u << 16);
}

#define WAIT_ASYNC(n) asm volatile("s_wait_asynccnt " #n ::: "memory")

// 16-byte global -> LDS async copy (VGLOBAL async encoding, ASYNCcnt-tracked).
// vDst = per-lane LDS byte address, vAddr = 64-bit global address, GV mode.
__device__ __forceinline__ void async_cp16(const ushort_t* g, ushort_t* l) {
    unsigned           lds = (unsigned)(unsigned long long)l;   // addr[31:0] = LDS byte addr
    unsigned long long ga  = (unsigned long long)g;
    asm volatile("global_load_async_to_lds_b128 %0, %1, off"
                 :: "v"(lds), "v"(ga) : "memory");
}

// ---------------------------------------------------------------------------
// Elementwise fp32 -> bf16 (4 elems/thread, packed 8B stores)
// ---------------------------------------------------------------------------
__global__ __launch_bounds__(256)
void cvt_bf16_kernel(const float* __restrict__ in, ushort_t* __restrict__ out) {
    size_t i = (size_t)blockIdx.x * 256 + threadIdx.x;
    float4 v = ((const float4*)in)[i];
    uint2 p;
    p.x = (unsigned)f2bf(v.x) | ((unsigned)f2bf(v.y) << 16);
    p.y = (unsigned)f2bf(v.z) | ((unsigned)f2bf(v.w) << 16);
    ((uint2*)out)[i] = p;
}

// ---------------------------------------------------------------------------
// fp32 W[K][N] -> bf16 Wt[N][K]   (32x32 LDS tile, both sides coalesced)
// ---------------------------------------------------------------------------
__global__ __launch_bounds__(256)
void cvt_transpose_kernel(const float* __restrict__ W, ushort_t* __restrict__ Wt,
                          int K, int N) {
    __shared__ ushort_t tile[32][33];
    const int tx = threadIdx.x & 31, ty = threadIdx.x >> 5;
    const int k0 = blockIdx.y * 32, n0 = blockIdx.x * 32;
#pragma unroll
    for (int i = 0; i < 4; ++i) {
        int k = ty + i * 8;
        tile[k][tx] = f2bf(W[(size_t)(k0 + k) * N + n0 + tx]);
    }
    __syncthreads();
#pragma unroll
    for (int i = 0; i < 4; ++i) {
        int n = ty + i * 8;
        Wt[(size_t)(n0 + n) * K + k0 + tx] = tile[tx][n];
    }
}

// ---------------------------------------------------------------------------
// GEMM: C[M,N] = A[M,K] @ B[K,N], A bf16 [M][K], B pre-transposed bf16 [N][K].
// 256 thr (8 waves), tile 128x128, K-step 32, double-buffered LDS filled with
// async global->LDS b128 copies (ASYNCcnt pipelining).
// ---------------------------------------------------------------------------
#define LDST 48   // halves; 96B rows keep every 16B chunk aligned

__device__ __forceinline__ void fill_tile(const ushort_t* __restrict__ A,
                                          const ushort_t* __restrict__ Bt,
                                          ushort_t* sA, ushort_t* sB,
                                          int m0, int n0, int K, int k0, int t) {
#pragma unroll
    for (int j = 0; j < 2; ++j) {
        int id  = t + j * 256;
        int row = id >> 2, c = (id & 3) * 8;
        async_cp16(A  + (size_t)(m0 + row) * K + k0 + c, sA + row * LDST + c);
        async_cp16(Bt + (size_t)(n0 + row) * K + k0 + c, sB + row * LDST + c);
    }
}

template <bool BF16_OUT>
__global__ __launch_bounds__(256)
void gemm_bf16_wmma(const ushort_t* __restrict__ A, const ushort_t* __restrict__ Bt,
                    void* __restrict__ Cv, int M, int N, int K) {
    __shared__ ushort_t sA[2][128 * LDST];
    __shared__ ushort_t sB[2][128 * LDST];

    const int t    = threadIdx.x;
    const int lane = t & 31;
    const int wave = t >> 5;
    const int half = lane >> 4;
    const int l16  = lane & 15;
    const int m0   = blockIdx.y * 128;
    const int n0   = blockIdx.x * 128;

    f32x8 acc[8];
#pragma unroll
    for (int i = 0; i < 8; ++i)
#pragma unroll
        for (int r = 0; r < 8; ++r) acc[i][r] = 0.0f;

    fill_tile(A, Bt, sA[0], sB[0], m0, n0, K, 0, t);

    const int nk = K >> 5;
    for (int i = 0; i < nk; ++i) {
        const int cur = i & 1;
        if (i + 1 < nk) {
            fill_tile(A, Bt, sA[cur ^ 1], sB[cur ^ 1], m0, n0, K, (i + 1) << 5, t);
            WAIT_ASYNC(4);     // only the 4 just-issued copies may remain
        } else {
            WAIT_ASYNC(0);
        }
        __syncthreads();

        Frag16 a;
        {
            const ushort_t* p = &sA[cur][(wave * 16 + l16) * LDST + half * 8];
            a.q[0] = *(const u32x4*)(p);       // K = half*8 + e
            a.q[1] = *(const u32x4*)(p + 16);  // K = 16 + half*8 + (e-8)
        }
        Frag16 bfr[8];
#pragma unroll
        for (int fn = 0; fn < 8; ++fn) {
            const ushort_t* p = &sB[cur][(fn * 16 + l16) * LDST + half * 8];
            bfr[fn].q[0] = *(const u32x4*)(p); // K = half*8 + e
            bfr[fn].q[1] = *(const u32x4*)(p + 16);
        }
#pragma unroll
        for (int fn = 0; fn < 8; ++fn)
            acc[fn] = __builtin_amdgcn_wmma_f32_16x16x32_bf16(
                false, a.v, false, bfr[fn].v, (short)0, acc[fn], false, false);
        __syncthreads();
    }

#pragma unroll
    for (int fn = 0; fn < 8; ++fn)
#pragma unroll
        for (int r = 0; r < 8; ++r) {
            int row = m0 + wave * 16 + half * 8 + r;
            int col = n0 + fn * 16 + l16;
            if (BF16_OUT) ((ushort_t*)Cv)[(size_t)row * N + col] = f2bf(acc[fn][r]);
            else          ((float*)Cv)   [(size_t)row * N + col] = acc[fn][r];
        }
}

// ---------------------------------------------------------------------------
// RoPE + QKV split: qkv bf16 (B,S,3H) ->
//   Qb, Kb : bf16 [bh][s][d] (rotary applied);  Vt : bf16 [bh][d][s]
// ---------------------------------------------------------------------------
__global__ __launch_bounds__(256)
void rope_split_kernel(const int* __restrict__ positions,
                       const ushort_t* __restrict__ qkv,
                       ushort_t* __restrict__ Qb, ushort_t* __restrict__ Kb,
                       ushort_t* __restrict__ Vt) {
    int idx = blockIdx.x * 256 + threadIdx.x;
    int j = idx & 63;
    int h = (idx >> 6) & 31;
    int s = (idx >> 11) & 2047;
    int b = idx >> 22;

    const ushort_t* row = qkv + (size_t)(b * Ssz + s) * (3 * Hsz);
    float pos = (float)positions[b * Ssz + s];
    float inv = __expf(-0.1439115683121279f * (float)j);   // theta^(-j/64)
    float sn, cs;
    __sincosf(pos * inv, &sn, &cs);

    int base = h * Dsz;
    float q1 = bf2f(row[base + j]),           q2 = bf2f(row[base + 64 + j]);
    float k1 = bf2f(row[Hsz + base + j]),     k2 = bf2f(row[Hsz + base + 64 + j]);
    float v1 = bf2f(row[2 * Hsz + base + j]), v2 = bf2f(row[2 * Hsz + base + 64 + j]);

    size_t qk = ((size_t)(b * NHsz + h) * Ssz + s) * Dsz;
    Qb[qk + j]      = f2bf(q1 * cs - q2 * sn);
    Qb[qk + 64 + j] = f2bf(q2 * cs + q1 * sn);
    Kb[qk + j]      = f2bf(k1 * cs - k2 * sn);
    Kb[qk + 64 + j] = f2bf(k2 * cs + k1 * sn);

    size_t vo = (size_t)(b * NHsz + h) * Dsz * Ssz + s;
    Vt[vo + (size_t)j        * Ssz] = f2bf(v1);
    Vt[vo + (size_t)(j + 64) * Ssz] = f2bf(v2);
}

// ---------------------------------------------------------------------------
// Causal flash attention. Grid: (S/64, B*NH); 128 threads = 4 waves.
// Output bf16 (B,S,H) feeding the out-proj GEMM.
// ---------------------------------------------------------------------------
#define PST 48

__global__ __launch_bounds__(128)
void attn_kernel(const ushort_t* __restrict__ Qb, const ushort_t* __restrict__ Kb,
                 const ushort_t* __restrict__ Vt, ushort_t* __restrict__ O) {
    __shared__ ushort_t sP[4 * 16 * PST];

    const int t    = threadIdx.x;
    const int lane = t & 31;
    const int wave = t >> 5;
    const int half = lane >> 4;
    const int l16  = lane & 15;
    const int bh   = blockIdx.y;
    const int b    = bh >> 5;
    const int h    = bh & 31;
    const int q0   = blockIdx.x * 64 + wave * 16;

    const size_t qkBase = (size_t)bh * Ssz * Dsz;
    const size_t vBase  = (size_t)bh * Dsz * Ssz;

    Frag16 qf[4];
    {
        const ushort_t* qrow = Qb + qkBase + (size_t)(q0 + l16) * Dsz;
#pragma unroll
        for (int kk = 0; kk < 4; ++kk) {
            const ushort_t* p = qrow + kk * 32 + half * 8;
            qf[kk].q[0] = *(const u32x4*)(p);
            qf[kk].q[1] = *(const u32x4*)(p + 16);
        }
    }

    f32x8 acc[8];
#pragma unroll
    for (int f = 0; f < 8; ++f)
#pragma unroll
        for (int r = 0; r < 8; ++r) acc[f][r] = 0.0f;
    float m[8], l[8];
#pragma unroll
    for (int r = 0; r < 8; ++r) { m[r] = -1e38f; l[r] = 0.0f; }

    const float scale = 0.08838834764831845f;   // 1/sqrt(128)
    const int kend = q0 + 16;

    for (int kb = 0; kb < kend; kb += 32) {
        f32x8 s[2];
#pragma unroll
        for (int nh = 0; nh < 2; ++nh) {
#pragma unroll
            for (int r = 0; r < 8; ++r) s[nh][r] = 0.0f;
            Frag16 kfr[4];
#pragma unroll
            for (int kk = 0; kk < 4; ++kk) {
                const ushort_t* p = Kb + qkBase +
                    (size_t)(kb + nh * 16 + l16) * Dsz + kk * 32 + half * 8;
                kfr[kk].q[0] = *(const u32x4*)(p);
                kfr[kk].q[1] = *(const u32x4*)(p + 16);
            }
#pragma unroll
            for (int kk = 0; kk < 4; ++kk)
                s[nh] = __builtin_amdgcn_wmma_f32_16x16x32_bf16(
                    false, qf[kk].v, false, kfr[kk].v, (short)0, s[nh], false, false);
        }

#pragma unroll
        for (int r = 0; r < 8; ++r) {
            int row = q0 + half * 8 + r;
            int c0 = kb + l16, c1 = kb + 16 + l16;
            float v0 = s[0][r] * scale;
            float v1 = s[1][r] * scale;
            if (c0 > row) v0 = -1e30f;
            if (c1 > row) v1 = -1e30f;

            float red = fmaxf(v0, v1);
            red = fmaxf(red, __shfl_xor(red, 1));
            red = fmaxf(red, __shfl_xor(red, 2));
            red = fmaxf(red, __shfl_xor(red, 4));
            red = fmaxf(red, __shfl_xor(red, 8));
            float mn = fmaxf(m[r], red);

            float alpha = __expf(m[r] - mn);
            float p0 = __expf(v0 - mn);
            float p1 = __expf(v1 - mn);
            s[0][r] = p0; s[1][r] = p1;

            float rs = p0 + p1;
            rs += __shfl_xor(rs, 1);
            rs += __shfl_xor(rs, 2);
            rs += __shfl_xor(rs, 4);
            rs += __shfl_xor(rs, 8);
            l[r] = l[r] * alpha + rs;
            m[r] = mn;
#pragma unroll
            for (int f = 0; f < 8; ++f) acc[f][r] *= alpha;
        }

        // P (16x32) -> LDS (bf16) -> A-fragment
        ushort_t* pw = &sP[wave * 16 * PST];
#pragma unroll
        for (int r = 0; r < 8; ++r) {
            int row = half * 8 + r;
            pw[row * PST + l16]      = f2bf(s[0][r]);
            pw[row * PST + 16 + l16] = f2bf(s[1][r]);
        }
        asm volatile("s_wait_dscnt 0" ::: "memory");   // intra-wave LDS RAW
        Frag16 pf;
        {
            const ushort_t* p = pw + l16 * PST + half * 8;
            pf.q[0] = *(const u32x4*)(p);
            pf.q[1] = *(const u32x4*)(p + 16);
        }

        // O += P V  (V^T layout -> contiguous B-frags), 2 groups of 4
#pragma unroll
        for (int g = 0; g < 2; ++g) {
            Frag16 vfr[4];
#pragma unroll
            for (int fi = 0; fi < 4; ++fi) {
                int f = g * 4 + fi;
                const ushort_t* p = Vt + vBase +
                    (size_t)(f * 16 + l16) * Ssz + kb + half * 8;
                vfr[fi].q[0] = *(const u32x4*)(p);
                vfr[fi].q[1] = *(const u32x4*)(p + 16);
            }
#pragma unroll
            for (int fi = 0; fi < 4; ++fi)
                acc[g * 4 + fi] = __builtin_amdgcn_wmma_f32_16x16x32_bf16(
                    false, pf.v, false, vfr[fi].v, (short)0, acc[g * 4 + fi], false, false);
        }
    }

#pragma unroll
    for (int f = 0; f < 8; ++f)
#pragma unroll
        for (int r = 0; r < 8; ++r) {
            int srow = q0 + half * 8 + r;
            int col  = h * Dsz + f * 16 + l16;
            O[((size_t)b * Ssz + srow) * Hsz + col] = f2bf(acc[f][r] / l[r]);
        }
}

// ---------------------------------------------------------------------------
extern "C" void kernel_launch(void* const* d_in, const int* in_sizes, int n_in,
                              void* d_out, int out_size, void* d_ws, size_t ws_size,
                              hipStream_t stream) {
    (void)in_sizes; (void)n_in; (void)out_size; (void)ws_size;
    const int*   positions = (const int*)  d_in[0];
    const float* hs        = (const float*)d_in[1];
    const float* w_pack    = (const float*)d_in[2];
    const float* w_o       = (const float*)d_in[3];
    float*       out       = (float*)d_out;

    char* ws = (char*)d_ws;
    ushort_t* hs_bf   = (ushort_t*)(ws);                  //  33,554,432 B
    ushort_t* wpack_t = (ushort_t*)(ws + 33554432u);      // 100,663,296 B  [3H][H]
    ushort_t* wo_t    = (ushort_t*)(ws + 134217728u);     //  33,554,432 B  [H][H]
    ushort_t* qkv_bf  = (ushort_t*)(ws + 167772160u);     // 100,663,296 B  (B,S,3H)
    ushort_t* Qb      = (ushort_t*)(ws + 268435456u);     //  33,554,432 B
    ushort_t* Kb      = (ushort_t*)(ws + 301989888u);
    ushort_t* Vt      = (ushort_t*)(ws + 335544320u);
    ushort_t* att_bf  = hs_bf;   // alias: hs_bf dead after QKV GEMM

    const int M = Bsz * Ssz;     // 4096

    // 0) precision/layout prep
    cvt_bf16_kernel<<<(M * Hsz) / 1024, 256, 0, stream>>>(hs, hs_bf);
    cvt_transpose_kernel<<<dim3(3 * Hsz / 32, Hsz / 32), 256, 0, stream>>>(
        w_pack, wpack_t, Hsz, 3 * Hsz);
    cvt_transpose_kernel<<<dim3(Hsz / 32, Hsz / 32), 256, 0, stream>>>(
        w_o, wo_t, Hsz, Hsz);

    // 1) qkv = hs @ w_pack  (bf16 out)
    gemm_bf16_wmma<true><<<dim3(3 * Hsz / 128, M / 128), 256, 0, stream>>>(
        hs_bf, wpack_t, qkv_bf, M, 3 * Hsz, Hsz);

    // 2) rope + split
    rope_split_kernel<<<(Bsz * Ssz * NHsz * 64) / 256, 256, 0, stream>>>(
        positions, qkv_bf, Qb, Kb, Vt);

    // 3) causal flash attention (bf16 out)
    attn_kernel<<<dim3(Ssz / 64, Bsz * NHsz), 128, 0, stream>>>(Qb, Kb, Vt, att_bf);

    // 4) out = att @ w_o  (fp32 out)
    gemm_bf16_wmma<false><<<dim3(Hsz / 128, M / 128), 256, 0, stream>>>(
        att_bf, wo_t, out, M, Hsz, Hsz);
}
